// DMHA_3255585210402
// MI455X (gfx1250) — matrix operand order repaired and checked by reading, not compiled, not run on hardware
//
#include <hip/hip_runtime.h>
#include <hip/hip_bf16.h>

#define BI  2
#define TI  2048
#define DI  2048
#define HI  16
#define HDI 128

typedef __attribute__((ext_vector_type(16))) __bf16 v16bf;
typedef __attribute__((ext_vector_type(8)))  float  v8f;
typedef __attribute__((ext_vector_type(4)))  unsigned int u32x4;

// ---- WMMA helpers ------------------------------------------------------
// A-matrix 16x32 bf16 layout (ISA 7.12.2): lane half0 holds K {0..7,16..23},
// half1 holds K {8..15,24..31}; row = lane%16.
__device__ __forceinline__ v16bf load_a_frag(const __bf16* p, int half) {
  union { v16bf v; u32x4 q[2]; } f;
  f.q[0] = *(const u32x4*)(p + 8 * half);
  f.q[1] = *(const u32x4*)(p + 16 + 8 * half);
  return f.v;
}
// B-matrix 32x16 bf16 layout: lane half0 holds K {0..7,16..23}, half1 K
// {8..15,24..31} of column n = lane%16 (matches A-fragment K order).
__device__ __forceinline__ v16bf load_b_frag(const __bf16* p, int half) {
  union { v16bf v; u32x4 q[2]; } f;
  f.q[0] = *(const u32x4*)(p + 8 * half);
  f.q[1] = *(const u32x4*)(p + 16 + 8 * half);
  return f.v;
}
__device__ __forceinline__ v8f wmma_bf16(v16bf a, v16bf b, v8f c) {
  return __builtin_amdgcn_wmma_f32_16x16x32_bf16(false, a, false, b,
                                                 (short)0, c, false, false);
}

// ---- prep: x -> bf16, V^T = bf16(x * c) in [B,H,HD,T] ------------------
__global__ __launch_bounds__(256) void prep_x_v(const float* __restrict__ x,
                                                const float* __restrict__ ve,
                                                __bf16* __restrict__ Xbf,
                                                __bf16* __restrict__ Vt) {
  const size_t i = (size_t)blockIdx.x * 256 + threadIdx.x;
  const int d = (int)(i & (DI - 1));
  const size_t bt = i >> 11;
  const int t = (int)(bt & (TI - 1));
  const int b = (int)(bt >> 11);
  const float xv = x[i];
  Xbf[i] = (__bf16)xv;
  // TOPK==NVK==4 -> top_k selects all rows 0..3: constant scale vector.
  const float c = 2.f * (ve[d] + ve[DI + d] + ve[2 * DI + d] + ve[3 * DI + d]);
  const int h = d >> 7, hd = d & (HDI - 1);
  Vt[((size_t)(b * HI + h) * HDI + hd) * TI + t] = (__bf16)(xv * c);
}

__global__ __launch_bounds__(256) void conv_w(const float* __restrict__ a,
                                              const float* __restrict__ b,
                                              const float* __restrict__ c,
                                              __bf16* __restrict__ A,
                                              __bf16* __restrict__ B,
                                              __bf16* __restrict__ C) {
  const size_t i = (size_t)blockIdx.x * 256 + threadIdx.x;
  A[i] = (__bf16)a[i];
  B[i] = (__bf16)b[i];
  C[i] = (__bf16)c[i];
}

// ---- GEMM: Y = A(bf16,[M,K=2048]) @ W^T(bf16,[N,K]) + bias -------------
// MODE 0: write bf16 to [B,H,T,HD] (for Q/K). MODE 1: write f32 to [M,N].
template <int MODE>
__global__ __launch_bounds__(256) void gemm_wmma_kernel(
    const __bf16* __restrict__ A, const __bf16* __restrict__ W,
    const float* __restrict__ bias, void* __restrict__ out) {
  const int wave = threadIdx.x >> 5, lane = threadIdx.x & 31;
  const int half = lane >> 4, ln = lane & 15;
  const int wm = wave & 3, wn = wave >> 2;
  const int mbase = blockIdx.x * 128 + wm * 32;
  const int nbase = blockIdx.y * 128 + wn * 64;

  v8f acc[2][4];
#pragma unroll
  for (int i = 0; i < 2; ++i)
#pragma unroll
    for (int j = 0; j < 4; ++j)
      acc[i][j] = (v8f){0.f, 0.f, 0.f, 0.f, 0.f, 0.f, 0.f, 0.f};

  const __bf16* arow[2] = {A + (size_t)(mbase + ln) * DI,
                           A + (size_t)(mbase + 16 + ln) * DI};
  const __bf16* wrow[4];
#pragma unroll
  for (int j = 0; j < 4; ++j) wrow[j] = W + (size_t)(nbase + j * 16 + ln) * DI;

  for (int k0 = 0; k0 < DI; k0 += 32) {
    __builtin_prefetch(arow[0] + k0 + 256, 0, 0);
    __builtin_prefetch(wrow[0] + k0 + 256, 0, 0);
    v16bf af[2], bf[4];
    af[0] = load_a_frag(arow[0] + k0, half);
    af[1] = load_a_frag(arow[1] + k0, half);
#pragma unroll
    for (int j = 0; j < 4; ++j) bf[j] = load_b_frag(wrow[j] + k0, half);
#pragma unroll
    for (int i = 0; i < 2; ++i)
#pragma unroll
      for (int j = 0; j < 4; ++j)
        acc[i][j] = wmma_bf16(af[i], bf[j], acc[i][j]);
  }

#pragma unroll
  for (int i = 0; i < 2; ++i) {
#pragma unroll
    for (int j = 0; j < 4; ++j) {
      const int col = nbase + j * 16 + ln;
      const float bv = bias[col];
#pragma unroll
      for (int r = 0; r < 8; ++r) {
        const int row = mbase + i * 16 + r + 8 * half;  // C-layout row map
        const float v = acc[i][j][r] + bv;
        if (MODE == 0) {
          const int b = row >> 11, t = row & (TI - 1);
          const int h = col >> 7, hd = col & (HDI - 1);
          ((__bf16*)out)[((size_t)(b * HI + h) * TI + t) * HDI + hd] = (__bf16)v;
        } else {
          ((float*)out)[(size_t)row * DI + col] = v;
        }
      }
    }
  }
}

// ---- Flash attention: 8 waves/block, each wave owns 16 query rows ------
__global__ __launch_bounds__(256) void attn_kernel(
    const __bf16* __restrict__ Q, const __bf16* __restrict__ Kb,
    const __bf16* __restrict__ Vt, __bf16* __restrict__ O) {
  __shared__ __bf16 plds[8][16 * 32];
  const int wave = threadIdx.x >> 5, lane = threadIdx.x & 31;
  const int half = lane >> 4, ln = lane & 15;
  const int bh = blockIdx.y;
  const int qrow = blockIdx.x * 128 + wave * 16;
  const __bf16* Qp = Q + (size_t)bh * TI * HDI;
  const __bf16* Kp = Kb + (size_t)bh * TI * HDI;
  const __bf16* Vp = Vt + (size_t)bh * HDI * TI;
  __bf16* pl = plds[wave];

  v16bf qf[4];
#pragma unroll
  for (int i = 0; i < 4; ++i)
    qf[i] = load_a_frag(Qp + (size_t)(qrow + ln) * HDI + i * 32, half);

  v8f o[8];
  float m[8], l[8];
#pragma unroll
  for (int n = 0; n < 8; ++n)
    o[n] = (v8f){0.f, 0.f, 0.f, 0.f, 0.f, 0.f, 0.f, 0.f};
#pragma unroll
  for (int r = 0; r < 8; ++r) {
    m[r] = -__builtin_inff();
    l[r] = 0.f;
  }

  const float scale = 0.08838834764831845f;  // 1/sqrt(128)
  const int kend = qrow + 16;
  for (int k0 = 0; k0 < kend; k0 += 32) {
    // S = Q K^T for two 16-key tiles
    v8f s[2];
#pragma unroll
    for (int j = 0; j < 2; ++j) {
      s[j] = (v8f){0.f, 0.f, 0.f, 0.f, 0.f, 0.f, 0.f, 0.f};
      const int key = k0 + j * 16 + ln;
#pragma unroll
      for (int i = 0; i < 4; ++i) {
        v16bf kf = load_b_frag(Kp + (size_t)key * HDI + i * 32, half);
        s[j] = wmma_bf16(qf[i], kf, s[j]);
      }
    }
    // online softmax per row (row = qrow + r + 8*half, cols in lanes%16)
#pragma unroll
    for (int r = 0; r < 8; ++r) {
      const int row = qrow + r + 8 * half;
      float s0 = s[0][r] * scale, s1 = s[1][r] * scale;
      if (k0 + ln > row) s0 = -__builtin_inff();
      if (k0 + 16 + ln > row) s1 = -__builtin_inff();
      float t = fmaxf(s0, s1);
      t = fmaxf(t, __shfl_xor(t, 1, 32));
      t = fmaxf(t, __shfl_xor(t, 2, 32));
      t = fmaxf(t, __shfl_xor(t, 4, 32));
      t = fmaxf(t, __shfl_xor(t, 8, 32));
      const float mn = fmaxf(m[r], t);
      const float alpha = __expf(m[r] - mn);
      const float p0 = __expf(s0 - mn), p1 = __expf(s1 - mn);
      float rs = p0 + p1;
      rs += __shfl_xor(rs, 1, 32);
      rs += __shfl_xor(rs, 2, 32);
      rs += __shfl_xor(rs, 4, 32);
      rs += __shfl_xor(rs, 8, 32);
      l[r] = l[r] * alpha + rs;
      m[r] = mn;
#pragma unroll
      for (int n = 0; n < 8; ++n) o[n][r] *= alpha;
      // stage P (C-layout) into LDS row-major 16x32 for A-fragment reload
      pl[(r + 8 * half) * 32 + ln] = (__bf16)p0;
      pl[(r + 8 * half) * 32 + 16 + ln] = (__bf16)p1;
    }
    asm volatile("s_wait_dscnt 0" ::: "memory");
    union { v16bf v; u32x4 q[2]; } pf;
    pf.q[0] = *(const u32x4*)(pl + ln * 32 + 8 * half);
    pf.q[1] = *(const u32x4*)(pl + ln * 32 + 16 + 8 * half);
    // O += P @ V  (V stored [hd][t] so B-fragments are contiguous)
#pragma unroll
    for (int n = 0; n < 8; ++n) {
      const int hd = n * 16 + ln;
      v16bf vf = load_b_frag(Vp + (size_t)hd * TI + k0, half);
      o[n] = wmma_bf16(pf.v, vf, o[n]);
    }
  }

  const int b = bh >> 4, h = bh & 15;
#pragma unroll
  for (int r = 0; r < 8; ++r) {
    const float inv = 1.f / l[r];
    const int t = qrow + r + 8 * half;
    __bf16* orow = O + ((size_t)b * TI + t) * DI + h * HDI;
#pragma unroll
    for (int n = 0; n < 8; ++n) orow[n * 16 + ln] = (__bf16)(o[n][r] * inv);
  }
}

extern "C" void kernel_launch(void* const* d_in, const int* in_sizes, int n_in,
                              void* d_out, int out_size, void* d_ws,
                              size_t ws_size, hipStream_t stream) {
  (void)in_sizes; (void)n_in; (void)out_size; (void)ws_size;
  const float* x  = (const float*)d_in[0];
  const float* Wq = (const float*)d_in[1];
  const float* bq = (const float*)d_in[2];
  const float* Wk = (const float*)d_in[3];
  const float* bk = (const float*)d_in[4];
  // d_in[5..7] (Wvq, bvq, v_keys) are dead: top_k with k==NVK selects all rows.
  const float* ve = (const float*)d_in[8];
  const float* Wo = (const float*)d_in[9];
  const float* bo = (const float*)d_in[10];

  const size_t XTD = (size_t)BI * TI * DI;  // 8,388,608 elems
  const size_t WDD = (size_t)DI * DI;       // 4,194,304 elems
  __bf16* Xbf = (__bf16*)d_ws;
  __bf16* Qbf = Xbf + XTD;   // [B,H,T,HD]
  __bf16* Kbf = Qbf + XTD;   // [B,H,T,HD]
  __bf16* Vt  = Kbf + XTD;   // [B,H,HD,T]
  __bf16* Obf = Vt + XTD;    // [B,T,D]
  __bf16* Wqb = Obf + XTD;
  __bf16* Wkb = Wqb + WDD;
  __bf16* Wob = Wkb + WDD;

  prep_x_v<<<(unsigned)(XTD / 256), 256, 0, stream>>>(x, ve, Xbf, Vt);
  conv_w<<<(unsigned)(WDD / 256), 256, 0, stream>>>(Wq, Wk, Wo, Wqb, Wkb, Wob);

  dim3 gg(BI * TI / 128, DI / 128);  // (32, 16)
  gemm_wmma_kernel<0><<<gg, 256, 0, stream>>>(Xbf, Wqb, bq, (void*)Qbf);
  gemm_wmma_kernel<0><<<gg, 256, 0, stream>>>(Xbf, Wkb, bk, (void*)Kbf);

  attn_kernel<<<dim3(TI / 128, BI * HI), 256, 0, stream>>>(Qbf, Kbf, Vt, Obf);

  gemm_wmma_kernel<1><<<gg, 256, 0, stream>>>(Obf, Wob, bo, d_out);
}
